// YOLOLayer_28630251995371
// MI455X (gfx1250) — matrix operand order repaired and checked by reading, not compiled, not run on hardware
//
#include <hip/hip_runtime.h>

// Problem constants (match reference)
#define BB      32
#define CIN     512
#define HGT     38
#define WID     38
#define NOUT    255            // NA*(NC+5)
#define HW      (HGT * WID)    // 1444
#define M_TOTAL (BB * HW)      // 46208 (divisible by 16)
#define M_TILES (M_TOTAL / 16) // 2888
#define KTILES  (CIN / 32)     // 16
#define NTILES  16             // 255 -> 256 padded
#define LDS_K   (CIN + 8)      // padded A row stride (halves) -> conflict-free ds_load_b128
#define STRIDE_F 16.0f

typedef __attribute__((ext_vector_type(16))) __bf16 v16bf;
typedef __attribute__((ext_vector_type(8)))  float  v8f;
typedef __attribute__((ext_vector_type(4)))  float  f4;

__device__ __forceinline__ unsigned short f32_bf16(float f) {
    unsigned int u = __float_as_uint(f);
    u += 0x7FFFu + ((u >> 16) & 1u);      // round-to-nearest-even
    return (unsigned short)(u >> 16);
}

// ---------------------------------------------------------------------------
// Prep kernel: convert conv_w (255x512 fp32) ONCE into bf16 WMMA-B fragments.
// Fragment index f = (ntile*KTILES + kt)*32 + lane ; each fragment = 16 bf16
// (32 B), so one wave's (ntile,kt) fragment set = 32 lanes * 32 B = 1 KB
// contiguous -> perfectly coalesced b128 loads in the GEMM hot loop.
// B lane layout: n = ntile*16 + lane%16 ; lanes 0-15: K=k0..k0+15,
// lanes 16-31: K=k0+16..k0+31 (k0 = kt*32). n==255 pad column baked to zero.
// ---------------------------------------------------------------------------
__global__ __launch_bounds__(256)
void pack_weights(const float* __restrict__ w, unsigned short* __restrict__ wp) {
    const int t     = blockIdx.x * blockDim.x + threadIdx.x; // 0..8191
    const int lane  = t & 31;
    const int kt    = (t >> 5) & (KTILES - 1);
    const int ntile = t >> 9;
    const int n     = ntile * 16 + (lane & 15);
    const int k0    = kt * 32 + ((lane >> 4) * 16);
    unsigned short* dst = wp + (size_t)t * 16;
    if (n < NOUT) {
        const float* src = w + (size_t)n * CIN + k0;
#pragma unroll
        for (int i = 0; i < 16; ++i) dst[i] = f32_bf16(src[i]);
    } else {
#pragma unroll
        for (int i = 0; i < 16; ++i) dst[i] = 0;
    }
}

// Fused bias + YOLO decode of one 16x16 f32 accumulator tile into the LDS
// staging buffer (layout [m_local][n], stride NOUT floats -> matches global).
// C/D layout: lane holds column n; rows m = vgpr r (+8 for lanes 16-31).
__device__ __forceinline__ void decode_to_lds(float* __restrict__ sOut,
                                              const float* __restrict__ bias,
                                              const v8f& acc, int n, int mtile, int hiHalf) {
    if (n >= NOUT) return;                // phantom pad column
    const float bv = bias[n];
    const int   j  = n % 85;              // channel within anchor
    const int   na = n / 85;
    const float aw = (na == 0) ? 30.0f : ((na == 1) ? 62.0f : 59.0f);
    const float ah = (na == 0) ? 61.0f : ((na == 1) ? 45.0f : 119.0f);
    const int mlbase = hiHalf * 8;
#pragma unroll
    for (int r = 0; r < 8; ++r) {
        const int ml = mlbase + r;
        const int m  = mtile * 16 + ml;
        float v = acc[r] + bv;
        float o;
        if (j == 0) {
            const int ww = (m % HW) % WID;
            o = (1.0f / (1.0f + __expf(-v)) + (float)ww) * STRIDE_F;
        } else if (j == 1) {
            const int hh = (m % HW) / WID;
            o = (1.0f / (1.0f + __expf(-v)) + (float)hh) * STRIDE_F;
        } else if (j == 2) {
            o = __expf(v) * aw;
        } else if (j == 3) {
            o = __expf(v) * ah;
        } else {
            o = v;                        // objectness + class logits raw
        }
        sOut[ml * NOUT + n] = o;
    }
}

__global__ __launch_bounds__(256)
void yolo_head_wmma(const float* __restrict__ xin,
                    const unsigned short* __restrict__ wpack,
                    const float* __restrict__ bias,
                    float* __restrict__ out) {
    // One buffer, two lives: bf16 A tile (16*520*2 = 16640 B) during the GEMM,
    // then fp32 output stage (16*255*4 = 16320 B) for coalesced streaming-out.
    __shared__ __align__(16) unsigned char smem[16 * LDS_K * 2];
    unsigned short* sA   = reinterpret_cast<unsigned short*>(smem);
    float*          sOut = reinterpret_cast<float*>(smem);

    const int tid   = threadIdx.x;
    const int mtile = blockIdx.x;

    // ---- cooperative A-tile load: fp32 NCHW gather -> bf16 LDS [m][k] ----
    // xin is touched exactly once by the whole grid: non-temporal loads keep
    // L2 free for the reused weight fragments.
    {
        const int mloc = tid & 15;
        const int kr   = tid >> 4;                       // 0..15
        const int m    = mtile * 16 + mloc;
        const int bimg = m / HW;
        const int hw   = m - bimg * HW;
        const float* src = xin + (size_t)bimg * CIN * HW + hw;
        for (int k = kr; k < CIN; k += 16)
            sA[mloc * LDS_K + k] = f32_bf16(__builtin_nontemporal_load(&src[(size_t)k * HW]));
    }
    __syncthreads();

    const int lane   = tid & 31;
    const int wave   = tid >> 5;        // 0..7
    const int ln     = lane & 15;
    const int hiHalf = lane >> 4;       // 0 | 1
    const int n0 = wave * 16 + ln;      // ntile = wave      (n in 0..127)
    const int n1 = (wave + 8) * 16 + ln;// ntile = wave + 8  (n in 128..255)
    const int kbA = hiHalf * 16;        // A: upper half-wave holds K+16..K+31

    v8f acc0 = {};
    v8f acc1 = {};

    const unsigned short* aRow = &sA[ln * LDS_K + kbA]; // 16B-aligned

    // Pre-packed B fragments: frag f = (ntile*KTILES + kt)*32 + lane, 2 uint4 each.
    const uint4* wq0 = reinterpret_cast<const uint4*>(wpack) +
                       ((size_t)(wave * KTILES) * 32 + lane) * 2;
    const uint4* wq1 = reinterpret_cast<const uint4*>(wpack) +
                       ((size_t)((wave + 8) * KTILES) * 32 + lane) * 2;

#pragma unroll 4
    for (int kt = 0; kt < KTILES; ++kt) {
        const int k0 = kt * 32;
        // A fragment 16x32 bf16: halves [0..7] = K kb..kb+7, [8..15] = K kb+8..kb+15
        union { uint4 q[2]; v16bf v; } af, b0, b1;
        af.q[0] = *reinterpret_cast<const uint4*>(aRow + k0);
        af.q[1] = *reinterpret_cast<const uint4*>(aRow + k0 + 8);
        b0.q[0] = wq0[0];
        b0.q[1] = wq0[1];
        b1.q[0] = wq1[0];
        b1.q[1] = wq1[1];
        wq0 += 64;                       // advance one kt (32 lanes * 2 uint4)
        wq1 += 64;

        acc0 = __builtin_amdgcn_wmma_f32_16x16x32_bf16(
                   false, af.v, false, b0.v, (short)0, acc0, false, false);
        acc1 = __builtin_amdgcn_wmma_f32_16x16x32_bf16(
                   false, af.v, false, b1.v, (short)0, acc1, false, false);
    }

    // ---- epilogue: decode into LDS stage, then stream out coalesced ----
    __syncthreads();                     // all waves done reading sA
    decode_to_lds(sOut, bias, acc0, n0, mtile, hiHalf);
    decode_to_lds(sOut, bias, acc1, n1, mtile, hiHalf);
    __syncthreads();

    // Block's output tile is one contiguous region: 16 rows * 255 ch = 4080
    // floats = 1020 aligned f4's starting at byte offset mtile*16320 (16B mult).
    {
        const f4* s4 = reinterpret_cast<const f4*>(sOut);
        f4* o4 = reinterpret_cast<f4*>(out + (size_t)mtile * 16 * NOUT);
        for (int f = tid; f < (16 * NOUT) / 4; f += 256)
            __builtin_nontemporal_store(s4[f], &o4[f]);
    }
}

extern "C" void kernel_launch(void* const* d_in, const int* in_sizes, int n_in,
                              void* d_out, int out_size, void* d_ws, size_t ws_size,
                              hipStream_t stream) {
    (void)in_sizes; (void)n_in; (void)out_size; (void)ws_size;
    const float* xin    = (const float*)d_in[0];
    const float* conv_w = (const float*)d_in[1];
    const float* conv_b = (const float*)d_in[2];
    float* out = (float*)d_out;
    unsigned short* wpack = (unsigned short*)d_ws;   // 256 KB used

    // Stage 1: one-shot weight convert + swizzle into WMMA-B fragment layout.
    pack_weights<<<dim3((NTILES * KTILES * 32) / 256), dim3(256), 0, stream>>>(conv_w, wpack);
    // Stage 2: fused GEMM + YOLO decode + coalesced NT store.
    yolo_head_wmma<<<dim3(M_TILES), dim3(256), 0, stream>>>(xin, wpack, conv_b, out);
}
